// StaticLatentEmbedder_14929306321243
// MI455X (gfx1250) — 8-bit matrix operand order repaired and checked by reading, compiled, not run on hardware
//
#include <hip/hip_runtime.h>
#include <hip/hip_bf16.h>

typedef int  v8i __attribute__((ext_vector_type(8)));
typedef int  v4i __attribute__((ext_vector_type(4)));

#define NB   64
#define LSEQ 256
#define CIN  512
#define HOUT 1024
#define MTOT (NB * LSEQ)
#define WELEMS (HOUT * CIN)

#define WS_APK   0
#define WS_BPK   (WS_APK + MTOT * CIN)
#define WS_DEQ   (WS_BPK + HOUT * CIN)
#define WS_PART  (WS_DEQ + MTOT * 4)
#define WS_WSC   (WS_PART + 512 * 4)

__global__ __launch_bounds__(256)
void wsum1_kernel(const float* __restrict__ w, float* __restrict__ partials) {
    __shared__ float sm[256];
    const int tid = threadIdx.x;
    const int base = blockIdx.x * 1024;
    float s = 0.f;
#pragma unroll
    for (int i = 0; i < 4; ++i) s += fabsf(w[base + tid + i * 256]);
    sm[tid] = s;
    __syncthreads();
    for (int off = 128; off > 0; off >>= 1) {
        if (tid < off) sm[tid] += sm[tid + off];
        __syncthreads();
    }
    if (tid == 0) partials[blockIdx.x] = sm[0];
}

__global__ __launch_bounds__(256)
void wsum2_kernel(const float* __restrict__ partials, float* __restrict__ wscale) {
    __shared__ float sm[256];
    const int tid = threadIdx.x;
    sm[tid] = partials[tid] + partials[tid + 256];
    __syncthreads();
    for (int off = 128; off > 0; off >>= 1) {
        if (tid < off) sm[tid] += sm[tid + off];
        __syncthreads();
    }
    if (tid == 0) wscale[0] = fmaxf(sm[0] / (float)WELEMS, 1e-5f);
}

__global__ __launch_bounds__(256)
void wpack_kernel(const float* __restrict__ w, const float* __restrict__ wscale_p,
                  signed char* __restrict__ Bpk) {
    const float inv = 1.0f / wscale_p[0];
    const int wid  = blockIdx.x * 8 + (threadIdx.x >> 5);
    const int lane = threadIdx.x & 31;
    const int ht = wid >> 3;
    const int kt = wid & 7;
    const int g = lane >> 4;
    const int n = lane & 15;
    const int h = ht * 16 + n;
    int q[8];
#pragma unroll
    for (int v = 0; v < 8; ++v) {
        unsigned int dw = 0;
#pragma unroll
        for (int b = 0; b < 4; ++b) {
            const int k = ((v >> 1) << 4) + (g << 3) + ((v & 1) << 2) + b;
            const float val = w[h * CIN + kt * 64 + k];
            int qi = (int)rintf(val * inv);
            qi = qi < -1 ? -1 : (qi > 1 ? 1 : qi);
            dw |= ((unsigned int)(qi & 0xff)) << (8 * b);
        }
        q[v] = (int)dw;
    }
    v4i lo = {q[0], q[1], q[2], q[3]};
    v4i hi = {q[4], q[5], q[6], q[7]};
    v4i* outp = (v4i*)(Bpk + ((size_t)(ht * 8 + kt) * 32 + lane) * 32);
    outp[0] = lo;
    outp[1] = hi;
}

__global__ __launch_bounds__(256)
void xpack_kernel(const float* __restrict__ x, const float* __restrict__ wscale_p,
                  signed char* __restrict__ Apk, float* __restrict__ deq) {
    __shared__ float tile[16 * CIN];
    __shared__ float qs[16];
    const int tid = threadIdx.x;
    const int mt = blockIdx.x;

    const float4* src = (const float4*)(x + (size_t)mt * 16 * CIN);
    float4* dst = (float4*)tile;
#pragma unroll
    for (int i = 0; i < 16; ++i) dst[tid + i * 256] = src[tid + i * 256];
    __syncthreads();

    const int tk = tid >> 4;
    const int sub = tid & 15;
    float mx = 0.f;
#pragma unroll
    for (int j = 0; j < 32; ++j)
        mx = fmaxf(mx, fabsf(tile[tk * CIN + sub + j * 16]));
    mx = fmaxf(mx, __shfl_xor(mx, 1));
    mx = fmaxf(mx, __shfl_xor(mx, 2));
    mx = fmaxf(mx, __shfl_xor(mx, 4));
    mx = fmaxf(mx, __shfl_xor(mx, 8));
    if (sub == 0) {
        const float a = fmaxf(mx, 1e-5f);
        qs[tk] = 127.0f / a;
        deq[mt * 16 + tk] = a * (wscale_p[0] / 127.0f);
    }
    __syncthreads();

    const int kt = tid >> 5;
    const int lane = tid & 31;
    const int g = lane >> 4;
    const int m = lane & 15;
    const float s = qs[m];
    int q[8];
#pragma unroll
    for (int v = 0; v < 8; ++v) {
        unsigned int dw = 0;
#pragma unroll
        for (int b = 0; b < 4; ++b) {
            const int k = ((v >> 1) << 4) + (g << 3) + ((v & 1) << 2) + b;
            const float val = tile[m * CIN + kt * 64 + k];
            int qi = (int)rintf(val * s);
            qi = qi < -128 ? -128 : (qi > 127 ? 127 : qi);
            dw |= ((unsigned int)(qi & 0xff)) << (8 * b);
        }
        q[v] = (int)dw;
    }
    v4i lo = {q[0], q[1], q[2], q[3]};
    v4i hi = {q[4], q[5], q[6], q[7]};
    v4i* outp = (v4i*)(Apk + ((size_t)(mt * 8 + kt) * 32 + lane) * 32);
    outp[0] = lo;
    outp[1] = hi;
}

__global__ __launch_bounds__(256)
void gemm_kernel(const signed char* __restrict__ Apk, const signed char* __restrict__ Bpk,
                 const float* __restrict__ deq, const float* __restrict__ null_emb,
                 const int* __restrict__ mask, float* __restrict__ out) {
    const int tid = threadIdx.x;
    const int mb = blockIdx.x;
    const int hb = blockIdx.y;
    const int batch = mb >> 1;

    if (mask[batch] != 0) {
#pragma unroll 4
        for (int i = 0; i < 32; ++i) {
            const int idx = tid + i * 256;
            const int r = idx >> 6;
            const int hc = idx & 63;
            const int m = mb * 128 + r;
            const int h = hb * 64 + hc;
            out[(size_t)m * HOUT + h] = null_emb[(size_t)(m & (LSEQ - 1)) * HOUT + h];
        }
        return;
    }

    const int wave = tid >> 5;
    const int lane = tid & 31;
    const int m_tile = mb * 8 + wave;
    const int ht0 = hb * 4;

    const v8i* A8 = (const v8i*)Apk;
    const v8i* B8 = (const v8i*)Bpk;

    v8i acc[4];
    acc[0] = (v8i){};
    acc[1] = (v8i){};
    acc[2] = (v8i){};
    acc[3] = (v8i){};

#pragma unroll
    for (int kt = 0; kt < 8; ++kt) {
        const v8i a = A8[(size_t)(m_tile * 8 + kt) * 32 + lane];
        const v8i b0 = B8[(size_t)((ht0 + 0) * 8 + kt) * 32 + lane];
        const v8i b1 = B8[(size_t)((ht0 + 1) * 8 + kt) * 32 + lane];
        const v8i b2 = B8[(size_t)((ht0 + 2) * 8 + kt) * 32 + lane];
        const v8i b3 = B8[(size_t)((ht0 + 3) * 8 + kt) * 32 + lane];
        acc[0] = __builtin_amdgcn_wmma_i32_16x16x64_iu8(true, a, true, b0, acc[0], false, false);
        acc[1] = __builtin_amdgcn_wmma_i32_16x16x64_iu8(true, a, true, b1, acc[1], false, false);
        acc[2] = __builtin_amdgcn_wmma_i32_16x16x64_iu8(true, a, true, b2, acc[2], false, false);
        acc[3] = __builtin_amdgcn_wmma_i32_16x16x64_iu8(true, a, true, b3, acc[3], false, false);
    }

    const int g = lane >> 4;
    const int nl = lane & 15;
    float dqr[8];
#pragma unroll
    for (int r = 0; r < 8; ++r) dqr[r] = deq[m_tile * 16 + g * 8 + r];
#pragma unroll
    for (int j = 0; j < 4; ++j) {
        const int h = (ht0 + j) * 16 + nl;
#pragma unroll
        for (int r = 0; r < 8; ++r) {
            const int m = m_tile * 16 + g * 8 + r;
            out[(size_t)m * HOUT + h] = (float)acc[j][r] * dqr[r];
        }
    }
}

extern "C" void kernel_launch(void* const* d_in, const int* in_sizes, int n_in,
                              void* d_out, int out_size, void* d_ws, size_t ws_size,
                              hipStream_t stream) {
    const float* x       = (const float*)d_in[0];
    const float* w       = (const float*)d_in[1];
    const float* nullemb = (const float*)d_in[2];
    const int*   mask    = (const int*)d_in[3];
    float* out = (float*)d_out;

    char* ws = (char*)d_ws;
    signed char* Apk   = (signed char*)(ws + WS_APK);
    signed char* Bpk   = (signed char*)(ws + WS_BPK);
    float* deq         = (float*)(ws + WS_DEQ);
    float* partials    = (float*)(ws + WS_PART);
    float* wscale      = (float*)(ws + WS_WSC);

    wsum1_kernel<<<512, 256, 0, stream>>>(w, partials);
    wsum2_kernel<<<1, 256, 0, stream>>>(partials, wscale);
    wpack_kernel<<<64, 256, 0, stream>>>(w, wscale, Bpk);
    xpack_kernel<<<MTOT / 16, 256, 0, stream>>>(x, wscale, Apk, deq);

    dim3 grid(128, 16);
    gemm_kernel<<<grid, 256, 0, stream>>>(Apk, Bpk, deq, nullemb, mask, out);
}
